// GCN_80977313399685
// MI455X (gfx1250) — compile-verified, matrix operand order checked by reading, not run on hardware
//
#include <hip/hip_runtime.h>
#include <hip/hip_bf16.h>

typedef __attribute__((ext_vector_type(16))) _Float16 v16h;
typedef __attribute__((ext_vector_type(8)))  float    v8f;

#define OUT_DIM 64

// ---------------------------------------------------------------------------
// Utility: zero a float buffer (graph-capture safe replacement for memset)
// ---------------------------------------------------------------------------
__global__ void gcn_zero_f32(float4* __restrict__ p, int n4) {
    int i = blockIdx.x * blockDim.x + threadIdx.x;
    if (i < n4) p[i] = float4{0.0f, 0.0f, 0.0f, 0.0f};
}

// ---------------------------------------------------------------------------
// Degree accumulation via float atomics (L2 atomic units)
// ---------------------------------------------------------------------------
__global__ void gcn_degree(const int* __restrict__ src, const int* __restrict__ dst,
                           float* __restrict__ out_deg, float* __restrict__ in_deg,
                           int n_edges) {
    int e = blockIdx.x * blockDim.x + threadIdx.x;
    if (e < n_edges) {
        atomicAdd(&out_deg[src[e]], 1.0f);
        atomicAdd(&in_deg[dst[e]], 1.0f);
    }
}

// In-place: deg -> clip(deg,1)^-0.5
__global__ void gcn_norm(float* __restrict__ p, int n) {
    int i = blockIdx.x * blockDim.x + threadIdx.x;
    if (i < n) p[i] = rsqrtf(fmaxf(p[i], 1.0f));
}

// ---------------------------------------------------------------------------
// GEMM: out[m, 0:64] = (H[m, 0:KD] * rnorm[m]) @ W[KD, 64]
// One wave per 16-row node tile; 4x v_wmma_f32_16x16x32_f16 per K-step.
// A/B fragment layouts per CDNA5 ISA 7.12.2 (16-bit, wave32):
//  - A per lane = two contiguous 8-float K-spans -> 4x global_load_b128
//  - B from LDS staged transposed+f16 -> 2x ds_load_b128 per fragment
// ---------------------------------------------------------------------------
template <int KD>
__global__ __launch_bounds__(256)
void gcn_gemm(const float* __restrict__ H, const float* __restrict__ W,
              const float* __restrict__ rnorm, float* __restrict__ out,
              int n_nodes, int n_tiles) {
    // sWt[n][k] = (f16) W[k][n]   (transposed, K contiguous per column)
    __shared__ _Float16 sWt[OUT_DIM * KD];
    for (int idx = threadIdx.x; idx < OUT_DIM * KD; idx += 256) {
        const int n = idx / KD;
        const int k = idx - n * KD;
        sWt[idx] = (_Float16)W[k * OUT_DIM + n];
    }
    __syncthreads();

    const int wave = threadIdx.x >> 5;
    const int lane = threadIdx.x & 31;
    const int tile = blockIdx.x * 8 + wave;
    if (tile >= n_tiles) return;

    const int half = lane >> 4;          // K-half selector (ISA layout)
    const int lmod = lane & 15;          // M (for A) / N (for B) index
    int mrow = tile * 16 + lmod;
    if (mrow >= n_nodes) mrow = n_nodes - 1;   // clamp (masked on store)
    const float rn = rnorm[mrow];
    const float* __restrict__ hrow = H + (size_t)mrow * KD + 8 * half;

    v8f acc[4] = {v8f{}, v8f{}, v8f{}, v8f{}};

    for (int k0 = 0; k0 < KD; k0 += 32) {
        // A fragment: lane's 16 halves = spans [k0+8h, +8) and [k0+16+8h, +8)
        const float4 x0 = *reinterpret_cast<const float4*>(hrow + k0);
        const float4 x1 = *reinterpret_cast<const float4*>(hrow + k0 + 4);
        const float4 y0 = *reinterpret_cast<const float4*>(hrow + k0 + 16);
        const float4 y1 = *reinterpret_cast<const float4*>(hrow + k0 + 20);
        v16h a;
        a[0]  = (_Float16)(x0.x * rn);  a[1]  = (_Float16)(x0.y * rn);
        a[2]  = (_Float16)(x0.z * rn);  a[3]  = (_Float16)(x0.w * rn);
        a[4]  = (_Float16)(x1.x * rn);  a[5]  = (_Float16)(x1.y * rn);
        a[6]  = (_Float16)(x1.z * rn);  a[7]  = (_Float16)(x1.w * rn);
        a[8]  = (_Float16)(y0.x * rn);  a[9]  = (_Float16)(y0.y * rn);
        a[10] = (_Float16)(y0.z * rn);  a[11] = (_Float16)(y0.w * rn);
        a[12] = (_Float16)(y1.x * rn);  a[13] = (_Float16)(y1.y * rn);
        a[14] = (_Float16)(y1.z * rn);  a[15] = (_Float16)(y1.w * rn);

        // 4 N-tiles of 16 columns; B pairs are packed f16 words in LDS
#pragma unroll
        for (int nt = 0; nt < 4; ++nt) {
            const _Float16* __restrict__ wcol = sWt + (nt * 16 + lmod) * KD + k0;
            v16h b;
#pragma unroll
            for (int i = 0; i < 8; ++i) {
                const int kb = ((i < 4) ? (2 * i) : (16 + 2 * (i - 4))) + 8 * half;
                b[2 * i]     = wcol[kb];
                b[2 * i + 1] = wcol[kb + 1];
            }
            acc[nt] = __builtin_amdgcn_wmma_f32_16x16x32_f16(
                false, a, false, b, (short)0, acc[nt], false, false);
        }
    }

    // D layout: lane lmod = N, VGPR r = row (r + 8*half)
    float* __restrict__ obase =
        out + (size_t)tile * 16 * OUT_DIM + (8 * half) * OUT_DIM + lmod;
    if (tile * 16 + 16 <= n_nodes) {
        // Full tile: straight-line stores, compile-time immediate offsets
#pragma unroll
        for (int nt = 0; nt < 4; ++nt) {
#pragma unroll
            for (int r = 0; r < 8; ++r) {
                obase[r * OUT_DIM + nt * 16] = acc[nt][r];
            }
        }
    } else {
        // Partial tail tile: per-row guard
#pragma unroll
        for (int nt = 0; nt < 4; ++nt) {
#pragma unroll
            for (int r = 0; r < 8; ++r) {
                const int row = tile * 16 + r + 8 * half;
                if (row < n_nodes)
                    obase[r * OUT_DIM + nt * 16] = acc[nt][r];
            }
        }
    }
}

// ---------------------------------------------------------------------------
// Edge scatter: one wave per edge, each lane atomically adds 2 floats.
// Source rows (25.6 MB table) and destination rows stay L2-resident.
// ---------------------------------------------------------------------------
__global__ __launch_bounds__(256)
void gcn_scatter(const float* __restrict__ hw, const int* __restrict__ src,
                 const int* __restrict__ dst, float* __restrict__ agg,
                 int n_edges) {
    const int e    = (int)((blockIdx.x * 256u + threadIdx.x) >> 5);
    const int lane = threadIdx.x & 31;
    if (e >= n_edges) return;
    const int s = src[e];
    const int d = dst[e];
    const float* __restrict__ pin  = hw  + (size_t)s * OUT_DIM;
    float* __restrict__       pout = agg + (size_t)d * OUT_DIM;
    const float2 v = *reinterpret_cast<const float2*>(pin + 2 * lane);
    atomicAdd(pout + 2 * lane,     v.x);
    atomicAdd(pout + 2 * lane + 1, v.y);
}

// ---------------------------------------------------------------------------
// Epilogue: h = relu(agg * norm_dst[row] + b[col])  (float4-vectorized)
// ---------------------------------------------------------------------------
__global__ void gcn_finalize(const float4* __restrict__ agg,
                             const float* __restrict__ norm_dst,
                             const float4* __restrict__ bias,
                             float4* __restrict__ out, int n4) {
    const int i = blockIdx.x * blockDim.x + threadIdx.x;
    if (i >= n4) return;
    const int row  = i >> 4;            // 16 float4 per 64-wide row
    const int col4 = i & 15;
    const float nd = norm_dst[row];
    const float4 a = agg[i];
    const float4 b = bias[col4];
    float4 r;
    r.x = fmaxf(a.x * nd + b.x, 0.0f);
    r.y = fmaxf(a.y * nd + b.y, 0.0f);
    r.z = fmaxf(a.z * nd + b.z, 0.0f);
    r.w = fmaxf(a.w * nd + b.w, 0.0f);
    out[i] = r;
}

// ---------------------------------------------------------------------------
extern "C" void kernel_launch(void* const* d_in, const int* in_sizes, int n_in,
                              void* d_out, int out_size, void* d_ws, size_t ws_size,
                              hipStream_t stream) {
    const float* feat = (const float*)d_in[0];
    const float* W1   = (const float*)d_in[1];
    const float* b1   = (const float*)d_in[2];
    const float* W2   = (const float*)d_in[3];
    const float* b2   = (const float*)d_in[4];
    const float* W3   = (const float*)d_in[5];
    const float* b3   = (const float*)d_in[6];
    const int*   src  = (const int*)d_in[7];
    const int*   dst  = (const int*)d_in[8];

    const int n_nodes = in_sizes[0] / 128;   // feat is [N, 128]
    const int n_edges = in_sizes[7];
    const int n_feat  = n_nodes * OUT_DIM;
    const int n_feat4 = n_feat / 4;          // OUT_DIM=64 -> always divisible
    const int n_tiles = (n_nodes + 15) / 16;

    // Workspace carve-out (256-byte aligned slices)
    char* ws = (char*)d_ws;
    size_t off = 0;
    auto carve = [&](size_t bytes) -> float* {
        float* p = (float*)(ws + off);
        off += (bytes + 255) & ~(size_t)255;
        return p;
    };
    float* normS = carve((size_t)n_nodes * 4);           // out-deg -> norm_src
    float* normD = carve((size_t)n_nodes * 4);           // in-deg  -> norm_dst
    float* bufA  = carve((size_t)n_feat  * 4);           // hw (GEMM output)
    float* bufB  = carve((size_t)n_feat  * 4);           // layer activations
    float* agg   = carve((size_t)n_feat  * 4);           // scatter accumulator

    const int TB = 256;
    const int blkNode  = (n_nodes + TB - 1) / TB;
    const int blkEdge  = (n_edges + TB - 1) / TB;
    const int blkFeat4 = (n_feat4 + TB - 1) / TB;
    const int blkEdgeW = (n_edges + 7) / 8;              // 8 waves / block
    const int blkGemm  = (n_tiles + 7) / 8;

    // --- degree norms ---
    {
        const int nn4 = (n_nodes + 3) / 4;               // zero via float4 chunks
        gcn_zero_f32<<<(nn4 + TB - 1) / TB, TB, 0, stream>>>((float4*)normS, nn4);
        gcn_zero_f32<<<(nn4 + TB - 1) / TB, TB, 0, stream>>>((float4*)normD, nn4);
    }
    gcn_degree<<<blkEdge, TB, 0, stream>>>(src, dst, normS, normD, n_edges);
    gcn_norm<<<blkNode, TB, 0, stream>>>(normS, n_nodes);
    gcn_norm<<<blkNode, TB, 0, stream>>>(normD, n_nodes);

    // --- layer 1: feat [N,128] -> bufB [N,64] ---
    gcn_gemm<128><<<blkGemm, TB, 0, stream>>>(feat, W1, normS, bufA, n_nodes, n_tiles);
    gcn_zero_f32<<<blkFeat4, TB, 0, stream>>>((float4*)agg, n_feat4);
    gcn_scatter<<<blkEdgeW, TB, 0, stream>>>(bufA, src, dst, agg, n_edges);
    gcn_finalize<<<blkFeat4, TB, 0, stream>>>((const float4*)agg, normD,
                                              (const float4*)b1, (float4*)bufB, n_feat4);

    // --- layer 2: bufB -> bufB (via bufA/agg) ---
    gcn_gemm<64><<<blkGemm, TB, 0, stream>>>(bufB, W2, normS, bufA, n_nodes, n_tiles);
    gcn_zero_f32<<<blkFeat4, TB, 0, stream>>>((float4*)agg, n_feat4);
    gcn_scatter<<<blkEdgeW, TB, 0, stream>>>(bufA, src, dst, agg, n_edges);
    gcn_finalize<<<blkFeat4, TB, 0, stream>>>((const float4*)agg, normD,
                                              (const float4*)b2, (float4*)bufB, n_feat4);

    // --- layer 3: bufB -> d_out ---
    gcn_gemm<64><<<blkGemm, TB, 0, stream>>>(bufB, W3, normS, bufA, n_nodes, n_tiles);
    gcn_zero_f32<<<blkFeat4, TB, 0, stream>>>((float4*)agg, n_feat4);
    gcn_scatter<<<blkEdgeW, TB, 0, stream>>>(bufA, src, dst, agg, n_edges);
    gcn_finalize<<<blkFeat4, TB, 0, stream>>>((const float4*)agg, normD,
                                              (const float4*)b3, (float4*)d_out, n_feat4);
}
